// BertAttentionMem_61770219651176
// MI455X (gfx1250) — matrix operand order repaired and checked by reading, not compiled, not run on hardware
//
#include <hip/hip_runtime.h>

// ---------------------------------------------------------------------------
// BERT attention block for gfx1250 (MI455X), bf16 WMMA + fp32 accumulate.
// Pipeline: pack(fp32->bf16) -> QKV GEMMs (WMMA) -> flash attention (WMMA)
//           -> output GEMM + residual (WMMA) -> LayerNorm.
// All matmul work goes through v_wmma_f32_16x16x32_bf16.
// ---------------------------------------------------------------------------

typedef __bf16 bhalf;
typedef __attribute__((ext_vector_type(16))) __bf16 v16bf;
typedef __attribute__((ext_vector_type(8)))  __bf16 bf16x8;
typedef __attribute__((ext_vector_type(4)))  __bf16 bf16x4;
typedef __attribute__((ext_vector_type(8)))  float  v8f;

union V16 { v16bf v; bf16x8 h[2]; };

#define HID   1024
#define NTOK  8192   // B*S = 8*1024
#define SEQ   1024
#define NHEAD 16
#define HDIM  64

static __device__ __forceinline__ v8f wmma_bf16(const V16& a, const V16& b, v8f c) {
  return __builtin_amdgcn_wmma_f32_16x16x32_bf16(
      /*neg_a=*/false, a.v, /*neg_b=*/false, b.v,
      /*c_mod=*/(short)0, c, /*reuse_a=*/false, /*reuse_b=*/false);
}

// ---------------------------------------------------------------- fp32->bf16
__global__ __launch_bounds__(256) void cvt_f32_bf16(const float* __restrict__ in,
                                                    bhalf* __restrict__ out, int n) {
  int i = (blockIdx.x * 256 + threadIdx.x) * 4;
  if (i < n) {
    float4 f = *(const float4*)(in + i);
    bf16x4 o;
    o[0] = (bhalf)f.x; o[1] = (bhalf)f.y; o[2] = (bhalf)f.z; o[3] = (bhalf)f.w;
    *(bf16x4*)(out + i) = o;
  }
}

// ------------------------------------------------------------ WMMA GEMM
// C[M,N] = A[M,K] * W[N,K]^T (+bias) (+resid). 128x128 block, BK=32,
// 256 threads = 8 waves; wave tile 32x64 = 2x4 wmma accumulators.
__global__ __launch_bounds__(256)
void gemm_bf16_wmma(const bhalf* __restrict__ A,    // [M][K]
                    const bhalf* __restrict__ W,    // [N][K] (torch layout)
                    const float* __restrict__ bias, // [N]
                    const float* __restrict__ resid,// [M][N] or null
                    bhalf* __restrict__ outb,       // [M][N] or null
                    float* __restrict__ outf,       // [M][N] or null
                    int M, int N, int K)
{
  const int LDT = 40;                 // 32 + 8 pad (80B rows, 16B aligned)
  __shared__ bhalf As[128 * 40];
  __shared__ bhalf Bs[128 * 40];

  const int tid  = threadIdx.x;
  const int lane = tid & 31, wid = tid >> 5;
  const int wm = wid & 3, wn = wid >> 2;
  const int m0 = blockIdx.y * 128, n0 = blockIdx.x * 128;
  const int rl = lane & 15, hi = lane >> 4;

  v8f acc[2][4];
  {
    v8f z = {};
    for (int i = 0; i < 2; ++i)
      for (int j = 0; j < 4; ++j) acc[i][j] = z;
  }

  const int ldr = tid >> 1;            // 0..127
  const int ldc = (tid & 1) * 16;      // 0 / 16

  for (int kt = 0; kt < K; kt += 32) {
    __syncthreads();
    const bhalf* ga = A + (size_t)(m0 + ldr) * K + kt + ldc;
    *(bf16x8*)&As[ldr * LDT + ldc]     = *(const bf16x8*)(ga);
    *(bf16x8*)&As[ldr * LDT + ldc + 8] = *(const bf16x8*)(ga + 8);
    const bhalf* gw = W + (size_t)(n0 + ldr) * K + kt + ldc;
    *(bf16x8*)&Bs[ldr * LDT + ldc]     = *(const bf16x8*)(gw);
    *(bf16x8*)&Bs[ldr * LDT + ldc + 8] = *(const bf16x8*)(gw + 8);
    if (kt + 32 < K) {                 // global_prefetch_b8 of next k-tile
      __builtin_prefetch(ga + 32, 0, 1);
      __builtin_prefetch(gw + 32, 0, 1);
    }
    __syncthreads();

    V16 afr[2];
    for (int i = 0; i < 2; ++i) {
      int row = wm * 32 + i * 16 + rl;
      afr[i].h[0] = *(const bf16x8*)&As[row * LDT + hi * 8];
      afr[i].h[1] = *(const bf16x8*)&As[row * LDT + hi * 8 + 16];
    }
    V16 bfr[4];
    for (int j = 0; j < 4; ++j) {
      int row = wn * 64 + j * 16 + rl;
      bfr[j].h[0] = *(const bf16x8*)&Bs[row * LDT + hi * 8];
      bfr[j].h[1] = *(const bf16x8*)&Bs[row * LDT + hi * 8 + 16];
    }
    for (int i = 0; i < 2; ++i)
      for (int j = 0; j < 4; ++j)
        acc[i][j] = wmma_bf16(afr[i], bfr[j], acc[i][j]);
  }

  for (int j = 0; j < 4; ++j) {
    int col = n0 + wn * 64 + j * 16 + rl;
    float bv = bias ? bias[col] : 0.0f;
    for (int i = 0; i < 2; ++i) {
      int row0 = m0 + wm * 32 + i * 16 + hi * 8;
      for (int rr = 0; rr < 8; ++rr) {
        size_t idx = (size_t)(row0 + rr) * N + col;
        float v = acc[i][j][rr] + bv;
        if (resid) v += resid[idx];
        if (outf) outf[idx] = v;
        else      outb[idx] = (bhalf)v;
      }
    }
  }
}

// ------------------------------------------------- flash attention (per head)
// block = 128 thr = 4 waves; each wave owns a 16-query strip of a 64-query
// tile; streams 32-key KV tiles with online softmax; all matmuls via WMMA.
__global__ __launch_bounds__(128)
void flash_attn(const bhalf* __restrict__ Qb, const bhalf* __restrict__ Kb,
                const bhalf* __restrict__ Vb, const float* __restrict__ mask,
                bhalf* __restrict__ Ctx)
{
  __shared__ bhalf Qs[64 * 72];       // [q][d]
  __shared__ bhalf Ks[32 * 72];       // [key][d]
  __shared__ bhalf Vt[64 * 40];       // [d][key] (transposed stage)
  __shared__ bhalf Ps[4 * 16 * 32];   // wave-private P slabs [q][key]

  const int tid = threadIdx.x, lane = tid & 31, wid = tid >> 5;
  const int b = blockIdx.x >> 4, nh = blockIdx.x & 15;
  const int q0 = blockIdx.y * 64, hc = nh * HDIM;
  const size_t rowbase = (size_t)b * SEQ;
  const int rl = lane & 15, hi = lane >> 4;

  { // load Q tile 64x64
    int r = tid >> 1, c0 = (tid & 1) * 32;
    const bhalf* gq = Qb + (rowbase + q0 + r) * HID + hc + c0;
    for (int u = 0; u < 4; ++u)
      *(bf16x8*)&Qs[r * 72 + c0 + u * 8] = *(const bf16x8*)(gq + u * 8);
  }
  __syncthreads();

  V16 aq[2];                          // Q A-frags, kept in registers
  for (int s = 0; s < 2; ++s) {
    int row = wid * 16 + rl;
    aq[s].h[0] = *(const bf16x8*)&Qs[row * 72 + s * 32 + hi * 8];
    aq[s].h[1] = *(const bf16x8*)&Qs[row * 72 + s * 32 + hi * 8 + 16];
  }

  float mrun[8], lrun[8];
  v8f oacc[4];
  {
    v8f z = {};
    for (int j = 0; j < 4; ++j) oacc[j] = z;
    for (int r = 0; r < 8; ++r) { mrun[r] = -1e30f; lrun[r] = 0.0f; }
  }

  for (int kt = 0; kt < SEQ; kt += 32) {
    __syncthreads();
    { // stage K (natural) and V (transposed) tiles
      int key = tid >> 2, c0 = (tid & 3) * 16;
      const bhalf* gk = Kb + (rowbase + kt + key) * HID + hc + c0;
      *(bf16x8*)&Ks[key * 72 + c0]     = *(const bf16x8*)gk;
      *(bf16x8*)&Ks[key * 72 + c0 + 8] = *(const bf16x8*)(gk + 8);
      const bhalf* gv = Vb + (rowbase + kt + key) * HID + hc + c0;
      bf16x8 v0 = *(const bf16x8*)gv;
      bf16x8 v1 = *(const bf16x8*)(gv + 8);
      for (int u = 0; u < 8; ++u) {
        Vt[(c0 + u) * 40 + key]     = v0[u];
        Vt[(c0 + 8 + u) * 40 + key] = v1[u];
      }
    }
    __syncthreads();

    // ---- scores: 16x32 per wave as two 16x16 WMMA tiles, K-dim 64 = 2 steps
    v8f sc[2];
    for (int t = 0; t < 2; ++t) {
      V16 bk[2];
      for (int s = 0; s < 2; ++s) {
        int row = t * 16 + rl;
        bk[s].h[0] = *(const bf16x8*)&Ks[row * 72 + s * 32 + hi * 8];
        bk[s].h[1] = *(const bf16x8*)&Ks[row * 72 + s * 32 + hi * 8 + 16];
      }
      v8f z = {};
      z = wmma_bf16(aq[0], bk[0], z);
      sc[t] = wmma_bf16(aq[1], bk[1], z);
    }

    const float mk0 = mask[(size_t)b * SEQ + kt + rl];
    const float mk1 = mask[(size_t)b * SEQ + kt + 16 + rl];

    float alpha[8];
    for (int rr = 0; rr < 8; ++rr) {
      float s0 = sc[0][rr] * 0.125f + mk0;
      float s1 = sc[1][rr] * 0.125f + mk1;
      float mx = fmaxf(s0, s1);
      for (int off = 1; off < 16; off <<= 1)            // 16-lane row groups
        mx = fmaxf(mx, __shfl_xor(mx, off, 32));
      float mnew = fmaxf(mrun[rr], mx);
      float al = __expf(mrun[rr] - mnew);
      float p0 = __expf(s0 - mnew);
      float p1 = __expf(s1 - mnew);
      float sum = p0 + p1;
      for (int off = 1; off < 16; off <<= 1)
        sum += __shfl_xor(sum, off, 32);
      lrun[rr] = lrun[rr] * al + sum;
      mrun[rr] = mnew;
      alpha[rr] = al;
      int prow = wid * 16 + rr + 8 * hi;
      Ps[prow * 32 + rl]      = (bhalf)p0;              // wave-private slab
      Ps[prow * 32 + 16 + rl] = (bhalf)p1;
    }
    for (int j = 0; j < 4; ++j)
      for (int rr = 0; rr < 8; ++rr) oacc[j][rr] *= alpha[rr];

    asm volatile("s_wait_dscnt 0" ::: "memory");        // LDS store->load turn

    V16 ap;
    {
      int prow = wid * 16 + rl;
      ap.h[0] = *(const bf16x8*)&Ps[prow * 32 + hi * 8];
      ap.h[1] = *(const bf16x8*)&Ps[prow * 32 + hi * 8 + 16];
    }
    for (int j = 0; j < 4; ++j) {                       // PV: one wmma / d-tile
      V16 bv;
      int vrow = j * 16 + rl;
      bv.h[0] = *(const bf16x8*)&Vt[vrow * 40 + hi * 8];
      bv.h[1] = *(const bf16x8*)&Vt[vrow * 40 + hi * 8 + 16];
      oacc[j] = wmma_bf16(ap, bv, oacc[j]);
    }
  }

  for (int j = 0; j < 4; ++j)
    for (int rr = 0; rr < 8; ++rr) {
      int qrow = q0 + wid * 16 + rr + 8 * hi;
      float inv = 1.0f / lrun[rr];
      Ctx[(rowbase + qrow) * HID + hc + j * 16 + rl] = (bhalf)(oacc[j][rr] * inv);
    }
}

// ---------------------------------------------------------------- LayerNorm
__global__ __launch_bounds__(256)
void layernorm_k(const float* __restrict__ X, const float* __restrict__ gamma,
                 const float* __restrict__ beta, float* __restrict__ out)
{
  __shared__ float ws1[8], ws2[8];
  const int row = blockIdx.x, tid = threadIdx.x;
  const int lane = tid & 31, wid = tid >> 5;
  const float* x = X + (size_t)row * HID;
  float v[4], s = 0.f, s2 = 0.f;
  for (int u = 0; u < 4; ++u) {
    float f = x[tid + u * 256];
    v[u] = f; s += f; s2 += f * f;
  }
  for (int off = 1; off < 32; off <<= 1) {              // wave32 reduce
    s  += __shfl_xor(s,  off, 32);
    s2 += __shfl_xor(s2, off, 32);
  }
  if (lane == 0) { ws1[wid] = s; ws2[wid] = s2; }
  __syncthreads();
  s = 0.f; s2 = 0.f;
  for (int w = 0; w < 8; ++w) { s += ws1[w]; s2 += ws2[w]; }
  const float mu = s * (1.0f / HID);
  const float var = s2 * (1.0f / HID) - mu * mu;
  const float rs = rsqrtf(var + 1e-12f);
  for (int u = 0; u < 4; ++u) {
    int c = tid + u * 256;
    out[(size_t)row * HID + c] = (v[u] - mu) * rs * gamma[c] + beta[c];
  }
}

// ---------------------------------------------------------------------------
extern "C" void kernel_launch(void* const* d_in, const int* in_sizes, int n_in,
                              void* d_out, int out_size, void* d_ws, size_t ws_size,
                              hipStream_t stream)
{
  const float* x    = (const float*)d_in[0];
  const float* mask = (const float*)d_in[1];
  const float* Wq = (const float*)d_in[2];  const float* bq = (const float*)d_in[3];
  const float* Wk = (const float*)d_in[4];  const float* bk = (const float*)d_in[5];
  const float* Wv = (const float*)d_in[6];  const float* bv = (const float*)d_in[7];
  const float* Wo = (const float*)d_in[8];  const float* bo = (const float*)d_in[9];
  const float* ga = (const float*)d_in[10]; const float* be = (const float*)d_in[11];

  // workspace carve-up (~126 MB total; intermediates stay L2-resident)
  char* ws = (char*)d_ws;
  size_t off = 0;
  auto carve = [&](size_t bytes) {
    void* p = ws + off;
    off += (bytes + 255) & ~(size_t)255;
    return p;
  };
  bhalf* Xb   = (bhalf*)carve((size_t)NTOK * HID * 2);
  bhalf* Wqb  = (bhalf*)carve((size_t)HID * HID * 2);
  bhalf* Wkb  = (bhalf*)carve((size_t)HID * HID * 2);
  bhalf* Wvb  = (bhalf*)carve((size_t)HID * HID * 2);
  bhalf* Wob  = (bhalf*)carve((size_t)HID * HID * 2);
  bhalf* Qb   = (bhalf*)carve((size_t)NTOK * HID * 2);
  bhalf* Kb   = (bhalf*)carve((size_t)NTOK * HID * 2);
  bhalf* Vb   = (bhalf*)carve((size_t)NTOK * HID * 2);
  bhalf* Ctxb = (bhalf*)carve((size_t)NTOK * HID * 2);
  float* Res  = (float*)carve((size_t)NTOK * HID * 4);

  // 1) pack fp32 -> bf16
  {
    int n = NTOK * HID;
    cvt_f32_bf16<<<(n / 4 + 255) / 256, 256, 0, stream>>>(x, Xb, n);
    int m = HID * HID;
    dim3 g((m / 4 + 255) / 256);
    cvt_f32_bf16<<<g, 256, 0, stream>>>(Wq, Wqb, m);
    cvt_f32_bf16<<<g, 256, 0, stream>>>(Wk, Wkb, m);
    cvt_f32_bf16<<<g, 256, 0, stream>>>(Wv, Wvb, m);
    cvt_f32_bf16<<<g, 256, 0, stream>>>(Wo, Wob, m);
  }

  // 2) Q/K/V projections (WMMA, bias fused, bf16 out)
  dim3 ggrid(HID / 128, NTOK / 128);
  gemm_bf16_wmma<<<ggrid, 256, 0, stream>>>(Xb, Wqb, bq, nullptr, Qb, nullptr, NTOK, HID, HID);
  gemm_bf16_wmma<<<ggrid, 256, 0, stream>>>(Xb, Wkb, bk, nullptr, Kb, nullptr, NTOK, HID, HID);
  gemm_bf16_wmma<<<ggrid, 256, 0, stream>>>(Xb, Wvb, bv, nullptr, Vb, nullptr, NTOK, HID, HID);

  // 3) flash attention (one block per (head, 64-query tile))
  flash_attn<<<dim3(8 * NHEAD, SEQ / 64), 128, 0, stream>>>(Qb, Kb, Vb, mask, Ctxb);

  // 4) output projection + residual (fp32 out)
  gemm_bf16_wmma<<<ggrid, 256, 0, stream>>>(Ctxb, Wob, bo, x, nullptr, Res, NTOK, HID, HID);

  // 5) LayerNorm -> d_out
  layernorm_k<<<NTOK, 256, 0, stream>>>(Res, ga, be, (float*)d_out);
}
